// OptimizedMultimodalModel_11785390260434
// MI455X (gfx1250) — compile-verified, matrix operand order checked by reading, not run on hardware
//
#include <hip/hip_runtime.h>
#include <hip/hip_bf16.h>
#include <hip/hip_fp16.h>

typedef __attribute__((ext_vector_type(16))) _Float16 v16h;
typedef __attribute__((ext_vector_type(8)))  _Float16 v8h;
typedef __attribute__((ext_vector_type(8)))  float    v8f;

#define EPSV 1e-5f

#define N_NODES 32768
#define N_EDGES 262144
#define BATCH   1024
#define IN_PAD  96        // 74 zero-padded to multiple of 32

// -------- order-preserving float<->uint key for atomic segment-max --------
__device__ __forceinline__ unsigned fkey(float f) {
  unsigned u = __float_as_uint(f);
  return (u & 0x80000000u) ? ~u : (u | 0x80000000u);
}
__device__ __forceinline__ float funkey(unsigned k) {
  unsigned u = (k & 0x80000000u) ? (k & 0x7FFFFFFFu) : ~k;
  return __uint_as_float(u);
}

// ---------------------------------------------------------------------------
// Tiled WMMA GEMM: out[m, colOff+n] = epilogue( A[M,K] @ B[K,N] )
//   REQUIRES: K % 32 == 0, M % 16 == 0, N % 16 == 0.
// Block = 4 waves; block tile = 64(M) x 16(N); each wave one 16x16 WMMA tile.
// B k-slab (32x16) is converted to f16 and staged transposed in LDS, shared
// by all 4 waves; A rows are loaded as aligned float4s straight from global.
// mode 0: raw f32 store              (GAT z = h @ W)
// mode 1: relu((acc+bias-bn_m)*bn_g*rsqrt(bn_v+eps)+bn_b)
// ---------------------------------------------------------------------------
__global__ void __launch_bounds__(128)
k_gemm_wmma(const float* __restrict__ A, const float* __restrict__ B,
            float* __restrict__ out, int M, int N, int K,
            int ldOut, int colOff, int mode,
            const float* __restrict__ bias,
            const float* __restrict__ bng, const float* __restrict__ bnb,
            const float* __restrict__ bnm, const float* __restrict__ bnv) {
  __shared__ __align__(16) _Float16 Bs[16 * 32];   // [col][row] halves

  const int tileN = blockIdx.x;
  const int tileM = blockIdx.y * 4 + threadIdx.y;
  const bool live = (tileM * 16) < M;

  const int lane = threadIdx.x;                // wave32
  const int half = lane >> 4;                  // frag K-chunk select
  const int l16  = lane & 15;
  const float* Ap = A + (long)(tileM * 16 + l16) * K;

  // cooperative B stage indices: 128 threads cover a 32x16 f32 slab
  const int tid  = threadIdx.y * 32 + threadIdx.x;
  const int bCol = tid & 15;                   // 0..15
  const int bRow = tid >> 4;                   // 0..7 (x4 groups of 8 rows)
  const float* Bp = B + (long)tileN * 16 + bCol;

  v8f acc = {};
  for (int k0 = 0; k0 < K; k0 += 32) {
    // ---- stage B[k0..k0+31][tileN*16..+15] -> LDS (transposed, f16) ----
#pragma unroll
    for (int q = 0; q < 4; ++q) {
      const int r = bRow + q * 8;
      Bs[bCol * 32 + r] = (_Float16)Bp[(long)(k0 + r) * N];
    }
    __syncthreads();

    // ---- A fragment: two aligned float4 pairs per lane ----
    const int kb = k0 + half * 8;
    v16h a, b;
    if (live) {
      const float4 x0 = *(const float4*)(Ap + kb);
      const float4 x1 = *(const float4*)(Ap + kb + 4);
      const float4 x2 = *(const float4*)(Ap + kb + 16);
      const float4 x3 = *(const float4*)(Ap + kb + 20);
      a[0]  = (_Float16)x0.x; a[1]  = (_Float16)x0.y; a[2]  = (_Float16)x0.z; a[3]  = (_Float16)x0.w;
      a[4]  = (_Float16)x1.x; a[5]  = (_Float16)x1.y; a[6]  = (_Float16)x1.z; a[7]  = (_Float16)x1.w;
      a[8]  = (_Float16)x2.x; a[9]  = (_Float16)x2.y; a[10] = (_Float16)x2.z; a[11] = (_Float16)x2.w;
      a[12] = (_Float16)x3.x; a[13] = (_Float16)x3.y; a[14] = (_Float16)x3.z; a[15] = (_Float16)x3.w;

      // ---- B fragment: two contiguous 128-bit LDS reads ----
      const v8h blo = *(const v8h*)&Bs[l16 * 32 + half * 8];
      const v8h bhi = *(const v8h*)&Bs[l16 * 32 + 16 + half * 8];
#pragma unroll
      for (int j = 0; j < 8; ++j) { b[j] = blo[j]; b[j + 8] = bhi[j]; }

      acc = __builtin_amdgcn_wmma_f32_16x16x32_f16(
          /*neg_a=*/false, a, /*neg_b=*/false, b,
          /*c_mod=*/(short)0, acc, /*reuse_a=*/false, /*reuse_b=*/false);
    }
    __syncthreads();
  }

  if (!live) return;
  const int n = tileN * 16 + l16;
  float scale = 1.f, shift = 0.f;
  if (mode == 1) {
    const float s = bng[n] * rsqrtf(bnv[n] + EPSV);
    scale = s;
    shift = (bias[n] - bnm[n]) * s + bnb[n];
  }
#pragma unroll
  for (int r = 0; r < 8; ++r) {
    const int m = tileM * 16 + half * 8 + r;   // C/D layout: lanes16-31 hold M+8
    float v = acc[r] * scale + shift;
    if (mode == 1) v = fmaxf(v, 0.f);
    out[(long)m * ldOut + colOff + n] = v;
  }
}

// -------- zero-pad node_feats [N,74] -> [N,96] --------
__global__ void k_pad(const float* __restrict__ src, float* __restrict__ dst, long total) {
  long t = (long)blockIdx.x * blockDim.x + threadIdx.x;
  if (t >= total) return;
  const int r = (int)(t / IN_PAD), c = (int)(t % IN_PAD);
  dst[t] = (c < 74) ? src[(long)r * 74 + c] : 0.f;
}

// -------- zero-fill scratch (graph-capture safe init) --------
__global__ void k_zero(unsigned* __restrict__ p, long n) {
  long t = (long)blockIdx.x * blockDim.x + threadIdx.x;
  if (t < n) p[t] = 0u;
}

// -------- GAT: el/er = einsum('nhd,hd->nh', z, al/ar) --------
__global__ void k_attn(const float* __restrict__ z, const float* __restrict__ al,
                       const float* __restrict__ ar, float* __restrict__ el,
                       float* __restrict__ er, int n) {
  int t = blockIdx.x * blockDim.x + threadIdx.x;
  if (t >= n * 4) return;
  const int h = t & 3;
  const float* zr = z + (long)(t >> 2) * 256 + h * 64;
  const float* a1 = al + h * 64;
  const float* a2 = ar + h * 64;
  float s1 = 0.f, s2 = 0.f;
#pragma unroll 8
  for (int d = 0; d < 64; ++d) { const float zz = zr[d]; s1 += zz * a1[d]; s2 += zz * a2[d]; }
  el[t] = s1; er[t] = s2;
}

// -------- GAT: e = leaky_relu(el[src]+er[dst]); segment-max via atomicMax --------
__global__ void k_edge_e(const float* __restrict__ el, const float* __restrict__ er,
                         const int* __restrict__ src, const int* __restrict__ dst,
                         float* __restrict__ ebuf, unsigned* __restrict__ mkey, int E) {
  int t = blockIdx.x * blockDim.x + threadIdx.x;
  if (t >= E * 4) return;
  const int e = t >> 2, h = t & 3;
  const float x  = el[src[e] * 4 + h] + er[dst[e] * 4 + h];
  const float lr = x > 0.f ? x : 0.2f * x;
  ebuf[t] = lr;
  atomicMax(&mkey[dst[e] * 4 + h], fkey(lr));
}

// -------- GAT: w = exp(e - m[dst]); den[dst] += w --------
__global__ void k_edge_w(const float* __restrict__ ebuf, const unsigned* __restrict__ mkey,
                         const int* __restrict__ dst, float* __restrict__ wbuf,
                         float* __restrict__ den, int E) {
  int t = blockIdx.x * blockDim.x + threadIdx.x;
  if (t >= E * 4) return;
  const int e = t >> 2, h = t & 3;
  const float m = funkey(mkey[dst[e] * 4 + h]);
  const float w = __expf(ebuf[t] - m);
  wbuf[t] = w;
  atomicAdd(&den[dst[e] * 4 + h], w);
}

// -------- GAT: out[dst] += (w/den[dst]) * z[src]   (per edge-feature) --------
__global__ void k_edge_agg(const float* __restrict__ wbuf, const float* __restrict__ den,
                           const int* __restrict__ src, const int* __restrict__ dst,
                           const float* __restrict__ z, float* __restrict__ out, long total) {
  long t = (long)blockIdx.x * blockDim.x + threadIdx.x;
  if (t >= total) return;
  const int e = (int)(t >> 8), f = (int)(t & 255), h = f >> 6;
  const int d = dst[e];
  const float alpha = wbuf[e * 4 + h] / den[d * 4 + h];
  atomicAdd(&out[(long)d * 256 + f], alpha * z[(long)src[e] * 256 + f]);
}

// -------- GAT: h = elu(out + bias)  (in place) --------
__global__ void k_elu_bias(float* __restrict__ h, const float* __restrict__ bias, long total) {
  long t = (long)blockIdx.x * blockDim.x + threadIdx.x;
  if (t >= total) return;
  const float x = h[t] + bias[(int)(t & 255)];
  h[t] = x > 0.f ? x : (__expf(x) - 1.f);
}

// -------- SumPooling: ge[node2graph[n]] += h[n] --------
__global__ void k_pool(const float* __restrict__ h, const int* __restrict__ n2g,
                       float* __restrict__ ge, long total) {
  long t = (long)blockIdx.x * blockDim.x + threadIdx.x;
  if (t >= total) return;
  const int nn = (int)(t >> 8), f = (int)(t & 255);
  atomicAdd(&ge[(long)n2g[nn] * 256 + f], h[t]);
}

// -------- final head: out = x @ W[256,1] + b --------
__global__ void k_fu3(const float* __restrict__ x, const float* __restrict__ w,
                      const float* __restrict__ b, float* __restrict__ out, int M) {
  int m = blockIdx.x * blockDim.x + threadIdx.x;
  if (m >= M) return;
  const float* xr = x + (long)m * 256;
  float s = 0.f;
#pragma unroll 8
  for (int k = 0; k < 256; ++k) s += xr[k] * w[k];
  out[m] = s + b[0];
}

// ---------------------------------------------------------------------------
static inline void gemm(const float* A, const float* B, float* out,
                        int M, int N, int K, int ldOut, int colOff, int mode,
                        const float* bias, const float* g, const float* bb,
                        const float* mm, const float* vv, hipStream_t s) {
  dim3 blk(32, 4);
  dim3 grd(N / 16, (M / 16 + 3) / 4);
  k_gemm_wmma<<<grd, blk, 0, s>>>(A, B, out, M, N, K, ldOut, colOff, mode,
                                  bias, g, bb, mm, vv);
}
static inline void zero(void* p, long words, hipStream_t s) {
  k_zero<<<(unsigned)((words + 255) / 256), 256, 0, s>>>((unsigned*)p, words);
}

extern "C" void kernel_launch(void* const* d_in, const int* in_sizes, int n_in,
                              void* d_out, int out_size, void* d_ws, size_t ws_size,
                              hipStream_t stream) {
  (void)in_sizes; (void)n_in; (void)out_size; (void)ws_size;
  // ---- inputs (setup_inputs insertion order, params dict recursively) ----
  const float* rdkit      = (const float*)d_in[0];   // [1024,2048]
  const float* chemberta  = (const float*)d_in[1];   // [1024,768]
  const float* node_feats = (const float*)d_in[2];   // [32768,74]
  int i = 3;
  const float* rd1W = (const float*)d_in[i++]; const float* rd1b = (const float*)d_in[i++];
  const float* rd1g = (const float*)d_in[i++]; const float* rd1bb = (const float*)d_in[i++];
  const float* rd1m = (const float*)d_in[i++]; const float* rd1v = (const float*)d_in[i++];
  const float* rd2W = (const float*)d_in[i++]; const float* rd2b = (const float*)d_in[i++];
  const float* rd2g = (const float*)d_in[i++]; const float* rd2bb = (const float*)d_in[i++];
  const float* rd2m = (const float*)d_in[i++]; const float* rd2v = (const float*)d_in[i++];
  const float* cbW  = (const float*)d_in[i++]; const float* cbb  = (const float*)d_in[i++];
  const float* cbg  = (const float*)d_in[i++]; const float* cbbb = (const float*)d_in[i++];
  const float* cbm  = (const float*)d_in[i++]; const float* cbv  = (const float*)d_in[i++];
  const float *gatW[3], *gatAL[3], *gatAR[3], *gatB[3];
  for (int l = 0; l < 3; ++l) {
    gatW[l]  = (const float*)d_in[i++]; gatAL[l] = (const float*)d_in[i++];
    gatAR[l] = (const float*)d_in[i++]; gatB[l]  = (const float*)d_in[i++];
  }
  const float* gpW  = (const float*)d_in[i++]; const float* gpb  = (const float*)d_in[i++];
  const float* gpg  = (const float*)d_in[i++]; const float* gpbb = (const float*)d_in[i++];
  const float* gpm  = (const float*)d_in[i++]; const float* gpv  = (const float*)d_in[i++];
  const float* f1W  = (const float*)d_in[i++]; const float* f1b  = (const float*)d_in[i++];
  const float* f1g  = (const float*)d_in[i++]; const float* f1bb = (const float*)d_in[i++];
  const float* f1m  = (const float*)d_in[i++]; const float* f1v  = (const float*)d_in[i++];
  const float* f2W  = (const float*)d_in[i++]; const float* f2b  = (const float*)d_in[i++];
  const float* f2g  = (const float*)d_in[i++]; const float* f2bb = (const float*)d_in[i++];
  const float* f2m  = (const float*)d_in[i++]; const float* f2v  = (const float*)d_in[i++];
  const float* f3W  = (const float*)d_in[i++]; const float* f3b  = (const float*)d_in[i++];
  const int* esrc = (const int*)d_in[i++];
  const int* edst = (const int*)d_in[i++];
  const int* n2g  = (const int*)d_in[i++];

  // ---- workspace carve ----
  char* base = (char*)d_ws;
  size_t off = 0;
  auto carve = [&](size_t bytes) -> char* {
    off = (off + 255) & ~(size_t)255;
    char* p = base + off;
    off += bytes;
    return p;
  };
  float* Z    = (float*)carve((size_t)N_NODES * 256 * 4);
  float* H0   = (float*)carve((size_t)N_NODES * 256 * 4);
  float* H1   = (float*)carve((size_t)N_NODES * 256 * 4);
  float* PAD  = (float*)carve((size_t)N_NODES * IN_PAD * 4);
  float* EL   = (float*)carve((size_t)N_NODES * 4 * 4);
  float* ER   = (float*)carve((size_t)N_NODES * 4 * 4);
  unsigned* MK = (unsigned*)carve((size_t)N_NODES * 4 * 4);
  float* DEN  = (float*)carve((size_t)N_NODES * 4 * 4);
  float* EB   = (float*)carve((size_t)N_EDGES * 4 * 4);
  float* WB   = (float*)carve((size_t)N_EDGES * 4 * 4);
  float* ACT1 = (float*)carve((size_t)BATCH * 256 * 4);
  float* FUSE = (float*)carve((size_t)BATCH * 512 * 4);
  float* GEMB = (float*)carve((size_t)BATCH * 256 * 4);
  float* FU1O = (float*)carve((size_t)BATCH * 512 * 4);
  float* FU2O = (float*)carve((size_t)BATCH * 256 * 4);

  // ---- rdkit branch: 2048->256->128 (writes fused cols [0,128)) ----
  gemm(rdkit, rd1W, ACT1, BATCH, 256, 2048, 256, 0, 1, rd1b, rd1g, rd1bb, rd1m, rd1v, stream);
  gemm(ACT1, rd2W, FUSE, BATCH, 128, 256, 512, 0, 1, rd2b, rd2g, rd2bb, rd2m, rd2v, stream);
  // ---- chemberta branch: 768->256 (fused cols [128,384)) ----
  gemm(chemberta, cbW, FUSE, BATCH, 256, 768, 512, 128, 1, cbb, cbg, cbbb, cbm, cbv, stream);

  // ---- pad node feats to K=96 so the WMMA GEMM needs no K guards ----
  {
    const long padTotal = (long)N_NODES * IN_PAD;
    k_pad<<<(unsigned)((padTotal + 255) / 256), 256, 0, stream>>>(node_feats, PAD, padTotal);
  }

  // ---- 3x GAT layers ----
  const float* hin = PAD;
  int Kin = IN_PAD;
  float* houts[3] = {H0, H1, H0};
  const long nodeFeatTotal = (long)N_NODES * 256;
  const long edgeTotal = (long)N_EDGES * 256;
  for (int l = 0; l < 3; ++l) {
    float* hout = houts[l];
    gemm(hin, gatW[l], Z, N_NODES, 256, Kin, 256, 0, 0,
         nullptr, nullptr, nullptr, nullptr, nullptr, stream);
    k_attn<<<(N_NODES * 4 + 255) / 256, 256, 0, stream>>>(Z, gatAL[l], gatAR[l], EL, ER, N_NODES);
    zero(MK, (long)N_NODES * 4, stream);
    zero(DEN, (long)N_NODES * 4, stream);
    k_edge_e<<<(N_EDGES * 4 + 255) / 256, 256, 0, stream>>>(EL, ER, esrc, edst, EB, MK, N_EDGES);
    k_edge_w<<<(N_EDGES * 4 + 255) / 256, 256, 0, stream>>>(EB, MK, edst, WB, DEN, N_EDGES);
    zero(hout, nodeFeatTotal, stream);
    k_edge_agg<<<(unsigned)((edgeTotal + 255) / 256), 256, 0, stream>>>(WB, DEN, esrc, edst, Z, hout, edgeTotal);
    k_elu_bias<<<(unsigned)((nodeFeatTotal + 255) / 256), 256, 0, stream>>>(hout, gatB[l], nodeFeatTotal);
    hin = hout;
    Kin = 256;
  }

  // ---- sum pooling + graph branch (fused cols [384,512)) ----
  zero(GEMB, (long)BATCH * 256, stream);
  k_pool<<<(unsigned)((nodeFeatTotal + 255) / 256), 256, 0, stream>>>(hin, n2g, GEMB, nodeFeatTotal);
  gemm(GEMB, gpW, FUSE, BATCH, 128, 256, 512, 384, 1, gpb, gpg, gpbb, gpm, gpv, stream);

  // ---- fusion head ----
  gemm(FUSE, f1W, FU1O, BATCH, 512, 512, 512, 0, 1, f1b, f1g, f1bb, f1m, f1v, stream);
  gemm(FU1O, f2W, FU2O, BATCH, 256, 512, 256, 0, 1, f2b, f2g, f2bb, f2m, f2v, stream);
  k_fu3<<<(BATCH + 255) / 256, 256, 0, stream>>>(FU2O, f3W, f3b, (float*)d_out, BATCH);
}
